// MultiResolutionDetector_6451040878891
// MI455X (gfx1250) — matrix operand order repaired and checked by reading, not compiled, not run on hardware
//
#include <hip/hip_runtime.h>

// ---------------------------------------------------------------------------
// MultiResolutionDetector for MI455X (gfx1250, wave32).
// 5x5 response conv runs on the matrix pipe via V_WMMA_F32_16X16X32_F16
// (banded-Toeplitz formulation: 5 accumulating WMMAs per 16x16 output tile).
//  - A operand: halo tile staged through LDS, assembled with 2x ds_load_b128.
//  - B operand: tile-invariant; pre-baked once per launch into a 5KB global
//    buffer in exact per-lane WMMA register order -> 2x global_load_b128.
// Everything else (resize / blur / NMS / key-sort top-k) is memory-bound VALU.
// ---------------------------------------------------------------------------

typedef __attribute__((ext_vector_type(16))) _Float16 v16h;
typedef __attribute__((ext_vector_type(8)))  _Float16 v8h;
typedef __attribute__((ext_vector_type(8)))  float    v8f;

#define NCAP   (1u << 18)          // per-level candidate capacity (pow2 for bitonic)
#define T_ALL  10432               // 1040+1560+1820+1950+2015+2047
#define N2     16384               // pow2 >= T_ALL for final sort
#define FILLV  (-1.7014117e38f)    // finfo(float32).min / 2

// ---- sortable key helpers --------------------------------------------------
__device__ __forceinline__ unsigned ordkey(float f) {
  unsigned u = __float_as_uint(f);
  return (u & 0x80000000u) ? ~u : (u | 0x80000000u);
}
__device__ __forceinline__ float invord(unsigned k) {
  unsigned u = (k & 0x80000000u) ? (k & 0x7FFFFFFFu) : ~k;
  return __uint_as_float(u);
}

// ---- bilinear resize (half-pixel centers, edge clamp; == jax bilinear) -----
__global__ void k_resize(const float* __restrict__ src, int sh, int sw,
                         float* __restrict__ dst, int dh, int dw) {
  int i = blockIdx.x * blockDim.x + threadIdx.x;
  if (i >= dh * dw) return;
  int y = i / dw, x = i % dw;
  float scy = (float)sh / (float)dh, scx = (float)sw / (float)dw;
  float fy = (y + 0.5f) * scy - 0.5f;
  float fx = (x + 0.5f) * scx - 0.5f;
  int y0 = (int)floorf(fy), x0 = (int)floorf(fx);
  float wy = fy - (float)y0, wx = fx - (float)x0;
  int y0c = min(max(y0, 0), sh - 1), y1c = min(max(y0 + 1, 0), sh - 1);
  int x0c = min(max(x0, 0), sw - 1), x1c = min(max(x0 + 1, 0), sw - 1);
  float v00 = src[y0c * sw + x0c], v01 = src[y0c * sw + x1c];
  float v10 = src[y1c * sw + x0c], v11 = src[y1c * sw + x1c];
  dst[i] = (1.f - wy) * ((1.f - wx) * v00 + wx * v01) +
           wy * ((1.f - wx) * v10 + wx * v11);
}

// ---- 5x5 gaussian blur (sigma=1, reflect border, separable -> outer prod) --
__device__ __forceinline__ int refl(int t, int n) {
  if (t < 0) t = -t;
  if (t >= n) t = 2 * n - 2 - t;
  return t;
}
__global__ void k_blur5(const float* __restrict__ src, float* __restrict__ dst,
                        int h, int w) {
  const float g[5] = {0.05448868f, 0.24420135f, 0.40261996f,
                      0.24420135f, 0.05448868f};
  int i = blockIdx.x * blockDim.x + threadIdx.x;
  if (i >= h * w) return;
  int y = i / w, x = i % w;
  float acc = 0.f;
#pragma unroll
  for (int iy = 0; iy < 5; ++iy) {
    int ry = refl(y + iy - 2, h);
    float racc = 0.f;
#pragma unroll
    for (int ix = 0; ix < 5; ++ix)
      racc += g[ix] * src[ry * w + refl(x + ix - 2, w)];
    acc += g[iy] * racc;
  }
  dst[i] = acc;
}

// ---- f32 -> f16 ------------------------------------------------------------
__global__ void k_tohalf(const float* __restrict__ src,
                         _Float16* __restrict__ dst, int n) {
  int i = blockIdx.x * blockDim.x + threadIdx.x;
  if (i < n) dst[i] = (_Float16)src[i];
}

// ---- pre-bake the 5 banded-Toeplitz B operands in WMMA register order ------
// bfmt[dy][lane][hh]:  lane = N + (K>=16 ? 16 : 0),  hh = K & 15,
// value = w[dy][K-N] if 0 <= K-N < 5 and K < 20, else 0.
__global__ void k_prep_b(const float* __restrict__ cw,
                         _Float16* __restrict__ bfmt) {
  int i = blockIdx.x * blockDim.x + threadIdx.x;   // 0 .. 5*32*16-1
  if (i >= 5 * 32 * 16) return;
  int dy   = i >> 9;           // /512
  int rem  = i & 511;
  int lane = rem >> 4;
  int hh   = rem & 15;
  int K = hh + ((lane >= 16) ? 16 : 0);
  int N = lane & 15;
  int t = K - N;
  float v = 0.f;
  if (K < 20 && t >= 0 && t < 5) v = cw[dy * 5 + t];
  bfmt[i] = (_Float16)v;
}

// ---- 5x5 conv response via WMMA -------------------------------------------
// One wave per 16x16 output tile.  D = sum_dy A_dy x B_dy + C where
//   A_dy[m][k] = img[y0-2+m+dy][x0-2+k]            (k = 0..19 meaningful)
//   B_dy[k][n] = w[dy][k-n] for 0 <= k-n < 5, k<20 (banded Toeplitz, else 0)
// => D[m][n] = SAME-padded 5x5 conv at (y0+m, x0+n).  f16 in, f32 accum.
// Halo tile (20 rows x 32 cols) staged via LDS; K>=20 slots of A carry finite
// clamped image data multiplied by B=0, so they contribute exactly 0.
__global__ void __launch_bounds__(32)
k_conv5_wmma(const _Float16* __restrict__ img, int h, int w,
             const _Float16* __restrict__ bfmt,
             const float* __restrict__ cb, float* __restrict__ score) {
  __shared__ _Float16 tile[20][32];   // 1280 B of LDS

  const int tx = blockIdx.x * 16;
  const int ty = blockIdx.y * 16;
  const int lane = threadIdx.x;       // 0..31, wave32
  const int m = lane & 15;            // A-row / B-col / D-col index
  const bool hi = lane >= 16;

  // ---- stage halo tile: rows ty-2..ty+17, cols tx-2..tx+29 (clamped) ------
  const int cx = tx - 2 + lane;
  const int gx = min(max(cx, 0), w - 1);
  const bool cin = (cx >= 0) && (cx < w);
#pragma unroll
  for (int r = 0; r < 20; ++r) {
    const int cy = ty - 2 + r;
    const int gy = min(max(cy, 0), h - 1);
    _Float16 v = img[(long)gy * w + gx];                 // unconditional load
    const bool rin = (cy >= 0) && (cy < h);
    tile[r][lane] = (rin && cin) ? v : (_Float16)0.f;    // branchless zero-fill
  }
  __syncthreads();   // single-wave workgroup: lowers to S_NOP; DS wait handles RAW

  const float bias = cb[0];

  v8f c = {};
#pragma unroll
  for (int dy = 0; dy < 5; ++dy) {
    // A: contiguous K order chosen here, same as bfmt. lane<16: halves 0..7 -> K=0..7, 8..15 -> K=8..15
    //                         lane>=16: halves 0..7 -> K=16..23, 8..15 -> K=24..31
    // Tile row for this lane = m + dy; the two K runs are contiguous 16B chunks.
    const v8h* rowp = (const v8h*)&tile[m + dy][0];      // chunks: K 0-7,8-15,16-23,24-31
    v8h p0 = rowp[hi ? 2 : 0];
    v8h p1 = rowp[hi ? 3 : 1];
    v16h a = __builtin_shufflevector(p0, p1, 0, 1, 2, 3, 4, 5, 6, 7,
                                     8, 9, 10, 11, 12, 13, 14, 15);
    // B: pre-baked per-lane register image, 32B per lane, coalesced + cached.
    const v8h* bp = (const v8h*)(bfmt + (size_t)(dy * 32 + lane) * 16);
    v16h b = __builtin_shufflevector(bp[0], bp[1], 0, 1, 2, 3, 4, 5, 6, 7,
                                     8, 9, 10, 11, 12, 13, 14, 15);
    c = __builtin_amdgcn_wmma_f32_16x16x32_f16(false, a, false, b, (short)0, c,
                                               false, false);
  }
  // D layout: VGPR r -> M = r + (hi?8:0), N = lane%16
#pragma unroll
  for (int r = 0; r < 8; ++r) {
    int oy = ty + r + (hi ? 8 : 0);
    int ox = tx + m;
    if (oy < h && ox < w) score[(long)oy * w + ox] = c[r] + bias;
  }
}

// ---- horizontal 15-max of border-masked score ------------------------------
__global__ void k_hmax(const float* __restrict__ s, float* __restrict__ o,
                       int h, int w) {
  int i = blockIdx.x * blockDim.x + threadIdx.x;
  if (i >= h * w) return;
  int y = i / w, x = i % w;
  bool rowin = (y >= 15 && y < h - 15);
  float mx = -3.4e38f;
  for (int dx = -7; dx <= 7; ++dx) {
    int q = x + dx;
    if (q < 0 || q >= w) continue;                       // OOB == -inf pad
    float v = (rowin && q >= 15 && q < w - 15) ? s[y * w + q] : 0.0f;
    mx = fmaxf(mx, v);
  }
  o[i] = mx;
}

// ---- candidate buffer init: k fill-entries (indices 0..k-1), rest zero -----
__global__ void k_cand_init(unsigned long long* __restrict__ cand, unsigned cap,
                            int kfill, unsigned* __restrict__ counter) {
  unsigned j = blockIdx.x * blockDim.x + threadIdx.x;
  if (j >= cap) return;
  unsigned long long key = 0ull;
  if ((int)j < kfill)
    key = ((unsigned long long)ordkey(FILLV) << 32) | (unsigned)(~j);
  cand[j] = key;
  if (j == 0) *counter = (unsigned)kfill;
}

// ---- NMS (vertical 15-max of hmax) + compaction ----------------------------
__global__ void k_compact(const float* __restrict__ s,
                          const float* __restrict__ hm, int h, int w,
                          unsigned long long* __restrict__ cand,
                          unsigned* __restrict__ counter, unsigned cap) {
  int i = blockIdx.x * blockDim.x + threadIdx.x;
  if (i >= h * w) return;
  int y = i / w, x = i % w;
  if (x < 15 || x >= w - 15 || y < 15 || y >= h - 15) return;  // border mask
  float v = s[i];
  if (!(v > 0.0f)) return;                                     // THR = 0
  float mx = -3.4e38f;
  for (int dy = -7; dy <= 7; ++dy) {
    int r = y + dy;
    if (r >= 0 && r < h) mx = fmaxf(mx, hm[r * w + x]);
  }
  if (v < mx) return;                                          // not window max
  unsigned long long key =
      ((unsigned long long)ordkey(v) << 32) | (unsigned)(~(unsigned)i);
  unsigned slot = atomicAdd(counter, 1u);
  if (slot < cap) cand[slot] = key;
}

// ---- global bitonic sort step (descending) ---------------------------------
__global__ void k_bitonic(unsigned long long* __restrict__ d, unsigned n,
                          unsigned j, unsigned kk) {
  unsigned i = blockIdx.x * blockDim.x + threadIdx.x;
  if (i >= n) return;
  unsigned ixj = i ^ j;
  if (ixj > i) {
    unsigned long long a = d[i], b = d[ixj];
    bool descBlock = ((i & kk) == 0);
    if (descBlock ? (a < b) : (a > b)) { d[i] = b; d[ixj] = a; }
  }
}

// ---- per-level emit of top-k into concatenated resp / laf arrays -----------
__global__ void k_emit_level(const unsigned long long* __restrict__ cand, int k,
                             int w, float f, int off,
                             float* __restrict__ resp_all,
                             float* __restrict__ laf_all) {
  int j = blockIdx.x * blockDim.x + threadIdx.x;
  if (j >= k) return;
  unsigned long long key = cand[j];
  float sc = invord((unsigned)(key >> 32));
  unsigned idx = ~(unsigned)(key & 0xFFFFFFFFu);
  resp_all[off + j] = sc;
  float xx = (float)(idx % (unsigned)w) * f;
  float yy = (float)(idx / (unsigned)w) * f;
  float ss = 22.0f * f;                     // 0.5*(f+f)*22
  float* L = laf_all + (size_t)(off + j) * 6;
  L[0] = ss; L[1] = 0.f; L[2] = xx;
  L[3] = 0.f; L[4] = ss; L[5] = yy;
}

// ---- final merge: build keys over all T candidates -------------------------
__global__ void k_build_keys(const float* __restrict__ resp, int T,
                             unsigned long long* __restrict__ keys, int n2) {
  int j = blockIdx.x * blockDim.x + threadIdx.x;
  if (j >= n2) return;
  keys[j] = (j < T) ? (((unsigned long long)ordkey(resp[j]) << 32) |
                       (unsigned)(~(unsigned)j))
                    : 0ull;
}

// ---- final emit: lafs then responses, in d_out -----------------------------
__global__ void k_emit_final(const unsigned long long* __restrict__ keys,
                             const float* __restrict__ laf_all,
                             float* __restrict__ out) {
  int j = blockIdx.x * blockDim.x + threadIdx.x;
  if (j >= 2048) return;
  unsigned long long key = keys[j];
  float sc = invord((unsigned)(key >> 32));
  unsigned idx = ~(unsigned)(key & 0xFFFFFFFFu);
  out[2048 * 6 + j] = sc;
  const float* L = laf_all + (size_t)idx * 6;
  float* O = out + (size_t)j * 6;
#pragma unroll
  for (int t = 0; t < 6; ++t) O[t] = L[t];
}

// ---------------------------------------------------------------------------
extern "C" void kernel_launch(void* const* d_in, const int* in_sizes, int n_in,
                              void* d_out, int out_size, void* d_ws,
                              size_t ws_size, hipStream_t stream) {
  (void)in_sizes; (void)n_in; (void)out_size; (void)ws_size;
  const float* img = (const float*)d_in[0];
  const float* cw  = (const float*)d_in[1];
  const float* cb  = (const float*)d_in[2];
  float* out = (float*)d_out;

  // ---- workspace layout ----
  char* ws = (char*)d_ws;
  size_t off = 0;
  auto alloc = [&](size_t bytes) -> char* {
    char* p = ws + off;
    off += (bytes + 255) & ~(size_t)255;
    return p;
  };
  unsigned* counter           = (unsigned*)alloc(256);
  _Float16* bfmt              = (_Float16*)alloc((size_t)5 * 32 * 16 * 2);
  unsigned long long* cand    = (unsigned long long*)alloc((size_t)NCAP * 8);
  float* resp_all             = (float*)alloc((size_t)T_ALL * 4);
  float* laf_all              = (float*)alloc((size_t)T_ALL * 24);
  unsigned long long* keys2   = (unsigned long long*)alloc((size_t)N2 * 8);
  float* imgup                = (float*)alloc((size_t)2896 * 2896 * 4);
  float* scorebuf             = (float*)alloc((size_t)2896 * 2896 * 4);
  float* hmaxbuf              = (float*)alloc((size_t)2896 * 2896 * 4);
  _Float16* img16             = (_Float16*)alloc((size_t)2896 * 2896 * 2);
  float* blurt                = (float*)alloc((size_t)2048 * 2048 * 4);
  float* pp0                  = (float*)alloc((size_t)1448 * 1448 * 4);
  float* pp1                  = (float*)alloc((size_t)1448 * 1448 * 4);

  auto sortbuf = [&](unsigned long long* buf, unsigned n) {
    unsigned blocks = (n + 255) / 256;
    for (unsigned kk = 2; kk <= n; kk <<= 1)
      for (unsigned j = kk >> 1; j >= 1; j >>= 1)
        k_bitonic<<<blocks, 256, 0, stream>>>(buf, n, j, kk);
  };

  auto detect = [&](const float* src, int n, float f, int k, int off_l) {
    int np = n * n;
    int b256 = (np + 255) / 256;
    k_tohalf<<<b256, 256, 0, stream>>>(src, img16, np);
    dim3 g((n + 15) / 16, (n + 15) / 16);
    k_conv5_wmma<<<g, 32, 0, stream>>>(img16, n, n, bfmt, cb, scorebuf);
    k_hmax<<<b256, 256, 0, stream>>>(scorebuf, hmaxbuf, n, n);
    k_cand_init<<<(NCAP + 255) / 256, 256, 0, stream>>>(cand, NCAP, k, counter);
    k_compact<<<b256, 256, 0, stream>>>(scorebuf, hmaxbuf, n, n, cand, counter,
                                        NCAP);
    sortbuf(cand, NCAP);
    k_emit_level<<<(k + 255) / 256, 256, 0, stream>>>(cand, k, n, f, off_l,
                                                      resp_all, laf_all);
  };

  // ---- bake WMMA B operands once ----
  k_prep_b<<<(5 * 32 * 16 + 255) / 256, 256, 0, stream>>>(cw, bfmt);

  // ---- level "up": resize 2048 -> 2896 (sqrt(2) upscale) ----
  k_resize<<<((2896 * 2896) + 255) / 256, 256, 0, stream>>>(img, 2048, 2048,
                                                            imgup, 2896, 2896);
  detect(imgup, 2896, 2048.0f / 2896.0f, 1040, 0);

  // ---- level 0: original resolution ----
  detect(img, 2048, 1.0f, 1560, 1040);

  // ---- pyramid levels 1..4 (gauss blur then bilinear downscale by sqrt2) ----
  k_blur5<<<((2048 * 2048) + 255) / 256, 256, 0, stream>>>(img, blurt, 2048, 2048);
  k_resize<<<((1448 * 1448) + 255) / 256, 256, 0, stream>>>(blurt, 2048, 2048,
                                                            pp0, 1448, 1448);
  detect(pp0, 1448, 2048.0f / 1448.0f, 1820, 2600);

  k_blur5<<<((1448 * 1448) + 255) / 256, 256, 0, stream>>>(pp0, blurt, 1448, 1448);
  k_resize<<<((1023 * 1023) + 255) / 256, 256, 0, stream>>>(blurt, 1448, 1448,
                                                            pp1, 1023, 1023);
  detect(pp1, 1023, 2048.0f / 1023.0f, 1950, 4420);

  k_blur5<<<((1023 * 1023) + 255) / 256, 256, 0, stream>>>(pp1, blurt, 1023, 1023);
  k_resize<<<((723 * 723) + 255) / 256, 256, 0, stream>>>(blurt, 1023, 1023,
                                                          pp0, 723, 723);
  detect(pp0, 723, 2048.0f / 723.0f, 2015, 6370);

  k_blur5<<<((723 * 723) + 255) / 256, 256, 0, stream>>>(pp0, blurt, 723, 723);
  k_resize<<<((511 * 511) + 255) / 256, 256, 0, stream>>>(blurt, 723, 723,
                                                          pp1, 511, 511);
  detect(pp1, 511, 2048.0f / 511.0f, 2047, 8385);

  // ---- final global top-2048 ----
  k_build_keys<<<(N2 + 255) / 256, 256, 0, stream>>>(resp_all, T_ALL, keys2, N2);
  sortbuf(keys2, N2);
  k_emit_final<<<(2048 + 255) / 256, 256, 0, stream>>>(keys2, laf_all, out);
}
